// H5Net_56401510531581
// MI455X (gfx1250) — matrix operand order repaired and checked by reading, not compiled, not run on hardware
//
#include <hip/hip_runtime.h>
#include <hip/hip_bf16.h>

typedef __attribute__((ext_vector_type(16))) _Float16 v16h;
typedef __attribute__((ext_vector_type(8)))  float    v8f;

#define NNODES 50000
#define NEDGES 800000
#define NGRAPHS 64

static __device__ __forceinline__ float silu_f(float x) {
  return x / (1.0f + __expf(-x));
}

// ---------------------------------------------------------------------------
// WMMA fragment helpers (V_WMMA_F32_16X16X32_F16, wave32)
//
// A 16x32 f16 fragment layout (per ISA 7.12.2): lane = r + 16*g holds row r,
// half i -> K = (i&7) + ((i>>3)<<4) + 8*g.
// B 32x16 f16 fragment: lane = n + 16*g holds col n, half i -> K = (i&7) + ((i>>3)<<4) + 8*g.
// C/D f32: lane = n + 16*g, vgpr j -> (M = j + 8*g, N = n).
//
// Activation tiles live in LDS in *A-fragment-swizzled* order and weights in
// global memory in *B-fragment-swizzled* order, so every fragment load is a
// contiguous 32-byte vector load (2 x b128) per lane.
// ---------------------------------------------------------------------------

// index of element (row r, col c) inside a swizzled 16 x (32*KC) f16 LDS tile
static __device__ __forceinline__ int swz_idx(int r, int c) {
  const int kc = c >> 5, kk = c & 31;
  const int lane = r + (((kk >> 3) & 1) << 4);
  const int i = (kk & 7) + ((kk >> 4) << 3);
  return ((kc << 5) + lane) * 16 + i;
}

static __device__ __forceinline__ v16h load_a_swz(const _Float16* S, int kc) {
  const int lane = threadIdx.x & 31;
  return *(const v16h*)(S + (((kc << 5) + lane) << 4));
}

// weights packed as [Kp/32][Np/16][32 lanes][16 halves]
static __device__ __forceinline__ v16h load_b_swz(const _Float16* W, int nt_total,
                                                  int kc, int nt) {
  const int lane = threadIdx.x & 31;
  return *(const v16h*)(W + ((((size_t)kc * nt_total + nt) << 5) + lane) * 16);
}

static __device__ __forceinline__ v8f wmma16(v16h a, v16h b, v8f c) {
  return __builtin_amdgcn_wmma_f32_16x16x32_f16(false, a, false, b, (short)0, c,
                                                false, false);
}

// ---------------------------------------------------------------------------
// Weight prep: f32 [K,N] row-major  ->  f16 B-fragment-swizzled, zero padded
// ---------------------------------------------------------------------------
__global__ void swz_convert(const float* __restrict__ src, _Float16* __restrict__ dst,
                            int K, int N, int Kp, int Np) {
  const int idx = blockIdx.x * blockDim.x + threadIdx.x;
  if (idx >= Kp * Np) return;
  const int i = idx & 15;
  const int lane = (idx >> 4) & 31;
  const int t = idx >> 9;
  const int NT = Np >> 4;
  const int nt = t % NT, kc = t / NT;
  const int k = kc * 32 + ((lane >> 4) << 3) + (i & 7) + ((i >> 3) << 4);
  const int n = nt * 16 + (lane & 15);
  dst[idx] = (k < K && n < N) ? (_Float16)src[(size_t)k * N + n] : (_Float16)0.f;
}

// ---------------------------------------------------------------------------
// Node embedding: feats0 = emb_table[z] @ emb_w + emb_b   (one thread / elem)
// ---------------------------------------------------------------------------
__global__ __launch_bounds__(256) void embed_kernel(const int* __restrict__ z,
                                                    const float* __restrict__ table,
                                                    const float* __restrict__ w,
                                                    const float* __restrict__ b,
                                                    float* __restrict__ out) {
  const int idx = blockIdx.x * blockDim.x + threadIdx.x;
  if (idx >= NNODES * 24) return;
  const int n = idx / 24, c = idx % 24;
  const float* e = table + z[n] * 24;
  float acc = b[c];
#pragma unroll
  for (int k = 0; k < 24; ++k) acc += e[k] * w[k * 24 + c];
  out[idx] = acc;
}

// ---------------------------------------------------------------------------
// Fused EGNN edge pipeline. One wave = 16 edges; all GEMMs via WMMA with
// LDS-resident intermediates. m_ij and coor_w*rel scattered with f32 atomics.
// ---------------------------------------------------------------------------
__global__ __launch_bounds__(128) void egnn_edge_kernel(
    const int* __restrict__ ei,            // [2, E]
    const float* __restrict__ feats,       // [N,24]
    const float* __restrict__ coors,       // [N,3]
    const _Float16* __restrict__ w1p,      // swz [64,128]
    const float* __restrict__ b1,          // [114]
    const _Float16* __restrict__ w2p,      // swz [128,64]
    const float* __restrict__ b2,          // [64]
    const _Float16* __restrict__ w3p,      // swz [64,256]
    const float* __restrict__ b3,          // [256]
    const float* __restrict__ w4,          // [256]
    const float* __restrict__ b4,          // [1]
    float* __restrict__ m_i,               // [N,64] accumulator (pre-zeroed)
    float* __restrict__ coors_out,         // [N,3] (pre-filled with coors_in)
    int update_coors) {
  __shared__ __attribute__((aligned(32))) _Float16 sX[4][16 * 64];   // e_in / m (KC=2)
  __shared__ __attribute__((aligned(32))) _Float16 sY[4][16 * 256];  // h1 / c1 (KC<=8)
  __shared__ float sRel[4][16 * 3];
  __shared__ int sDst[4][16];

  const int wave = threadIdx.x >> 5;
  const int lane = threadIdx.x & 31;
  const int g = lane >> 4;
  const int nl = lane & 15;
  const int e0 = (blockIdx.x * 4 + wave) * 16;

  _Float16* X = sX[wave];
  _Float16* Y = sY[wave];
  float* rel = sRel[wave];
  int* dstb = sDst[wave];

  // ---- gather + edge features (lanes 0..15, one edge each) ----
  if (lane < 16) {
    const int e = e0 + lane;
    const int s = ei[e];
    const int d = ei[NEDGES + e];
    dstb[lane] = d;
    const float rx = coors[s * 3 + 0] - coors[d * 3 + 0];
    const float ry = coors[s * 3 + 1] - coors[d * 3 + 1];
    const float rz = coors[s * 3 + 2] - coors[d * 3 + 2];
    rel[lane * 3 + 0] = rx; rel[lane * 3 + 1] = ry; rel[lane * 3 + 2] = rz;
    const float dd = rx * rx + ry * ry + rz * rz;
#pragma unroll
    for (int c = 0; c < 24; ++c) X[swz_idx(lane, c)] = (_Float16)feats[d * 24 + c];
#pragma unroll
    for (int c = 0; c < 24; ++c) X[swz_idx(lane, 24 + c)] = (_Float16)feats[s * 24 + c];
    float sc = 1.0f;
#pragma unroll
    for (int j = 0; j < 4; ++j) {
      const float v = dd * sc;
      X[swz_idx(lane, 48 + j)] = (_Float16)sinf(v);
      X[swz_idx(lane, 52 + j)] = (_Float16)cosf(v);
      sc *= 0.5f;
    }
    X[swz_idx(lane, 56)] = (_Float16)dd;
#pragma unroll
    for (int j = 57; j < 64; ++j) X[swz_idx(lane, j)] = (_Float16)0.f;
  }
  __syncthreads();

  // ---- GEMM1: [16x64(57)] x [64x128(114)] -> h1 (silu) in Y ----
  for (int nt = 0; nt < 8; ++nt) {
    v8f acc = {0.f, 0.f, 0.f, 0.f, 0.f, 0.f, 0.f, 0.f};
#pragma unroll
    for (int kc = 0; kc < 2; ++kc)
      acc = wmma16(load_a_swz(X, kc), load_b_swz(w1p, 8, kc, nt), acc);
    const int n = nt * 16 + nl;
    const float bias = (n < 114) ? b1[n] : 0.f;
#pragma unroll
    for (int j = 0; j < 8; ++j)
      Y[swz_idx(j + 8 * g, n)] = (_Float16)silu_f(acc[j] + bias);
  }
  __syncthreads();

  // ---- GEMM2: [16x128] x [128x64] -> m_ij (silu) in X ----
  for (int nt = 0; nt < 4; ++nt) {
    v8f acc = {0.f, 0.f, 0.f, 0.f, 0.f, 0.f, 0.f, 0.f};
#pragma unroll
    for (int kc = 0; kc < 4; ++kc)
      acc = wmma16(load_a_swz(Y, kc), load_b_swz(w2p, 4, kc, nt), acc);
    const int n = nt * 16 + nl;
    const float bias = b2[n];
#pragma unroll
    for (int j = 0; j < 8; ++j)
      X[swz_idx(j + 8 * g, n)] = (_Float16)silu_f(acc[j] + bias);
  }
  __syncthreads();

  // ---- scatter m_ij into m_i[dst] (L2-resident f32 atomics) ----
#pragma unroll 4
  for (int r = 0; r < 16; ++r) {
    const int d = dstb[r];
    unsafeAtomicAdd(&m_i[d * 64 + lane], (float)X[swz_idx(r, lane)]);
    unsafeAtomicAdd(&m_i[d * 64 + 32 + lane], (float)X[swz_idx(r, 32 + lane)]);
  }

  if (update_coors) {
    // ---- GEMM3: [16x64] x [64x256] -> c1 (silu) in Y ----
    for (int nt = 0; nt < 16; ++nt) {
      v8f acc = {0.f, 0.f, 0.f, 0.f, 0.f, 0.f, 0.f, 0.f};
#pragma unroll
      for (int kc = 0; kc < 2; ++kc)
        acc = wmma16(load_a_swz(X, kc), load_b_swz(w3p, 16, kc, nt), acc);
      const int n = nt * 16 + nl;
      const float bias = b3[n];
#pragma unroll
      for (int j = 0; j < 8; ++j)
        Y[swz_idx(j + 8 * g, n)] = (_Float16)silu_f(acc[j] + bias);
    }
    __syncthreads();

    // ---- coor_w = c1 . w4 + b4 ; scatter coor_w * rel into coors_out ----
    float s = 0.f;
    for (int j = 0; j < 128; ++j)
      s += (float)Y[swz_idx(nl, g * 128 + j)] * w4[g * 128 + j];
    s += __shfl_xor(s, 16, 32);
    if (lane < 16) {
      const float cw = s + b4[0];
      const int d = dstb[lane];
      unsafeAtomicAdd(&coors_out[d * 3 + 0], cw * rel[lane * 3 + 0]);
      unsafeAtomicAdd(&coors_out[d * 3 + 1], cw * rel[lane * 3 + 1]);
      unsafeAtomicAdd(&coors_out[d * 3 + 2], cw * rel[lane * 3 + 2]);
    }
  }
}

// ---------------------------------------------------------------------------
// Node update MLP: feats_out = feats + nw2 . silu(nw1 . [feats, m_i] + nb1) + nb2
// ---------------------------------------------------------------------------
__global__ __launch_bounds__(256) void node_kernel(
    const float* __restrict__ feats, const float* __restrict__ m_i,
    const float* __restrict__ w1, const float* __restrict__ b1,
    const float* __restrict__ w2, const float* __restrict__ b2,
    float* __restrict__ out) {
  const int n = blockIdx.x * blockDim.x + threadIdx.x;
  if (n >= NNODES) return;
  float h[48];
#pragma unroll
  for (int j = 0; j < 48; ++j) h[j] = b1[j];
  for (int i = 0; i < 88; ++i) {
    const float xi = (i < 24) ? feats[n * 24 + i] : m_i[n * 64 + (i - 24)];
    const float* wr = w1 + i * 48;
#pragma unroll
    for (int j = 0; j < 48; ++j) h[j] += xi * wr[j];
  }
#pragma unroll
  for (int j = 0; j < 48; ++j) h[j] = silu_f(h[j]);
  float o[24];
#pragma unroll
  for (int c = 0; c < 24; ++c) o[c] = b2[c];
  for (int j = 0; j < 48; ++j) {
    const float hj = h[j];
    const float* wr = w2 + j * 24;
#pragma unroll
    for (int c = 0; c < 24; ++c) o[c] += hj * wr[c];
  }
#pragma unroll
  for (int c = 0; c < 24; ++c) out[n * 24 + c] = feats[n * 24 + c] + o[c];
}

// ---------------------------------------------------------------------------
// Fused FFNN (72->256->256->256) + pooled segment-sum. One wave = 16 nodes.
// ---------------------------------------------------------------------------
__global__ __launch_bounds__(64) void ffnn_kernel(
    const float* __restrict__ f0, const float* __restrict__ f1,
    const float* __restrict__ f2, const int* __restrict__ bidx,
    const _Float16* __restrict__ w0p, const float* __restrict__ b0,
    const _Float16* __restrict__ w1p, const float* __restrict__ b1,
    const _Float16* __restrict__ w2p, const float* __restrict__ b2,
    float* __restrict__ pooled, int ntiles) {
  __shared__ __attribute__((aligned(32))) _Float16 sA[2][16 * 256];
  __shared__ __attribute__((aligned(32))) _Float16 sB[2][16 * 256];
  const int wave = threadIdx.x >> 5;
  const int lane = threadIdx.x & 31;
  const int g = lane >> 4, nl = lane & 15;
  const int tile_raw = blockIdx.x * 2 + wave;
  const bool live = tile_raw < ntiles;
  const int tile = live ? tile_raw : (ntiles - 1);
  const int n0 = tile * 16;
  _Float16* A = sA[wave];
  _Float16* B = sB[wave];

  // build x = silu(concat(f0,f1,f2)) padded to 96 cols, in A-fragment order
  if (lane < 16) {
    int node = n0 + lane;
    if (node >= NNODES) node = NNODES - 1;
#pragma unroll
    for (int c = 0; c < 24; ++c) {
      A[swz_idx(lane, c)]      = (_Float16)silu_f(f0[node * 24 + c]);
      A[swz_idx(lane, 24 + c)] = (_Float16)silu_f(f1[node * 24 + c]);
      A[swz_idx(lane, 48 + c)] = (_Float16)silu_f(f2[node * 24 + c]);
    }
#pragma unroll
    for (int c = 72; c < 96; ++c) A[swz_idx(lane, c)] = (_Float16)0.f;
  }
  __syncthreads();

  // layer0: [16x96(72)] x [96x256] -> B
  for (int nt = 0; nt < 16; ++nt) {
    v8f acc = {0.f, 0.f, 0.f, 0.f, 0.f, 0.f, 0.f, 0.f};
#pragma unroll
    for (int kc = 0; kc < 3; ++kc)
      acc = wmma16(load_a_swz(A, kc), load_b_swz(w0p, 16, kc, nt), acc);
    const int n = nt * 16 + nl;
    const float bias = b0[n];
#pragma unroll
    for (int j = 0; j < 8; ++j)
      B[swz_idx(j + 8 * g, n)] = (_Float16)silu_f(acc[j] + bias);
  }
  __syncthreads();

  // layer1: [16x256] x [256x256] -> A
  for (int nt = 0; nt < 16; ++nt) {
    v8f acc = {0.f, 0.f, 0.f, 0.f, 0.f, 0.f, 0.f, 0.f};
#pragma unroll
    for (int kc = 0; kc < 8; ++kc)
      acc = wmma16(load_a_swz(B, kc), load_b_swz(w1p, 16, kc, nt), acc);
    const int n = nt * 16 + nl;
    const float bias = b1[n];
#pragma unroll
    for (int j = 0; j < 8; ++j)
      A[swz_idx(j + 8 * g, n)] = (_Float16)silu_f(acc[j] + bias);
  }
  __syncthreads();

  // layer2 + pooled segment-sum scatter
  for (int nt = 0; nt < 16; ++nt) {
    v8f acc = {0.f, 0.f, 0.f, 0.f, 0.f, 0.f, 0.f, 0.f};
#pragma unroll
    for (int kc = 0; kc < 8; ++kc)
      acc = wmma16(load_a_swz(A, kc), load_b_swz(w2p, 16, kc, nt), acc);
    const int n = nt * 16 + nl;
    const float bias = b2[n];
#pragma unroll
    for (int j = 0; j < 8; ++j) {
      const int node = n0 + j + 8 * g;
      if (live && node < NNODES)
        unsafeAtomicAdd(&pooled[bidx[node] * 256 + n], silu_f(acc[j] + bias));
    }
  }
}

__global__ __launch_bounds__(256) void count_kernel(const int* __restrict__ bidx,
                                                    float* __restrict__ cnt) {
  const int n = blockIdx.x * blockDim.x + threadIdx.x;
  if (n < NNODES) unsafeAtomicAdd(&cnt[bidx[n]], 1.0f);
}

__global__ __launch_bounds__(256) void final_kernel(const float* __restrict__ pooled,
                                                    const float* __restrict__ cnt,
                                                    const float* __restrict__ fow,
                                                    const float* __restrict__ fob,
                                                    float* __restrict__ out) {
  const int t = threadIdx.x;  // 256 threads = 64 graphs x 4 outputs
  const int gph = t >> 2, o = t & 3;
  float acc = 0.f;
  for (int j = 0; j < 256; ++j) acc += pooled[gph * 256 + j] * fow[j * 4 + o];
  float c = cnt[gph];
  c = c > 1.f ? c : 1.f;
  acc = acc / c + fob[o];
  out[gph * 4 + o] = 1.f / (1.f + __expf(-acc));
}

// ---------------------------------------------------------------------------
extern "C" void kernel_launch(void* const* d_in, const int* in_sizes, int n_in,
                              void* d_out, int out_size, void* d_ws, size_t ws_size,
                              hipStream_t stream) {
  const int* z = (const int*)d_in[0];
  const float* pos = (const float*)d_in[1];
  const int* ei = (const int*)d_in[2];
  const int* bidx = (const int*)d_in[3];
  const float* emb_table = (const float*)d_in[4];
  const float* emb_w = (const float*)d_in[5];
  const float* emb_b = (const float*)d_in[6];
  const float* ew1 = (const float*)d_in[7];
  const float* eb1 = (const float*)d_in[8];
  const float* ew2 = (const float*)d_in[9];
  const float* eb2 = (const float*)d_in[10];
  const float* cw1 = (const float*)d_in[11];
  const float* cb1 = (const float*)d_in[12];
  const float* cw2 = (const float*)d_in[13];
  const float* cb2 = (const float*)d_in[14];
  const float* nw1 = (const float*)d_in[15];
  const float* nb1 = (const float*)d_in[16];
  const float* nw2 = (const float*)d_in[17];
  const float* nb2 = (const float*)d_in[18];
  const float* f0w = (const float*)d_in[19];
  const float* f0b = (const float*)d_in[20];
  const float* f1w = (const float*)d_in[21];
  const float* f1b = (const float*)d_in[22];
  const float* f2w = (const float*)d_in[23];
  const float* f2b = (const float*)d_in[24];
  const float* fow = (const float*)d_in[25];
  const float* fob = (const float*)d_in[26];

  char* ws = (char*)d_ws;
  size_t off = 0;
  auto walloc = [&](size_t bytes) -> char* {
    char* p = ws + off;
    off += (bytes + 255) & ~(size_t)255;
    return p;
  };
  float* feats0 = (float*)walloc((size_t)NNODES * 24 * 4);
  float* feats1 = (float*)walloc((size_t)NNODES * 24 * 4);
  float* feats2 = (float*)walloc((size_t)NNODES * 24 * 4);
  float* cB     = (float*)walloc((size_t)NNODES * 3 * 4);
  float* m_i    = (float*)walloc((size_t)NNODES * 64 * 4);
  float* pooled = (float*)walloc((size_t)NGRAPHS * 256 * 4);
  float* cnt    = (float*)walloc((size_t)NGRAPHS * 4);
  _Float16* ew1p = (_Float16*)walloc((size_t)2 * 64 * 128 * 2);
  _Float16* ew2p = (_Float16*)walloc((size_t)2 * 128 * 64 * 2);
  _Float16* cw1p = (_Float16*)walloc((size_t)2 * 64 * 256 * 2);
  _Float16* f0wp = (_Float16*)walloc((size_t)96 * 256 * 2);
  _Float16* f1wp = (_Float16*)walloc((size_t)256 * 256 * 2);
  _Float16* f2wp = (_Float16*)walloc((size_t)256 * 256 * 2);

  // accumulator init (harness does not re-zero between replays)
  hipMemsetAsync(m_i, 0, (size_t)NNODES * 64 * 4, stream);
  hipMemsetAsync(pooled, 0, (size_t)NGRAPHS * 256 * 4, stream);
  hipMemsetAsync(cnt, 0, (size_t)NGRAPHS * 4, stream);
  hipMemcpyAsync(cB, pos, (size_t)NNODES * 3 * 4, hipMemcpyDeviceToDevice, stream);

  // weight prep: convert + pad + swizzle into B-fragment order
  auto swz = [&](const float* s, _Float16* d, int K, int N, int Kp, int Np) {
    const int tot = Kp * Np;
    swz_convert<<<(tot + 255) / 256, 256, 0, stream>>>(s, d, K, N, Kp, Np);
  };
  for (int k = 0; k < 2; ++k) {
    swz(ew1 + k * 57 * 114, ew1p + k * 64 * 128, 57, 114, 64, 128);
    swz(ew2 + k * 114 * 64, ew2p + k * 128 * 64, 114, 64, 128, 64);
    swz(cw1 + k * 64 * 256, cw1p + k * 64 * 256, 64, 256, 64, 256);
  }
  swz(f0w, f0wp, 72, 256, 96, 256);
  swz(f1w, f1wp, 256, 256, 256, 256);
  swz(f2w, f2wp, 256, 256, 256, 256);

  embed_kernel<<<(NNODES * 24 + 255) / 256, 256, 0, stream>>>(z, emb_table, emb_w,
                                                              emb_b, feats0);

  // ---- EGNN layer 0 (coors update needed for layer 1) ----
  egnn_edge_kernel<<<NEDGES / 64, 128, 0, stream>>>(
      ei, feats0, pos, ew1p, eb1, ew2p, eb2, cw1p, cb1, cw2, cb2, m_i, cB, 1);
  node_kernel<<<(NNODES + 255) / 256, 256, 0, stream>>>(feats0, m_i, nw1, nb1, nw2,
                                                        nb2, feats1);

  // ---- EGNN layer 1 (final coors unused downstream -> skip coor MLP) ----
  hipMemsetAsync(m_i, 0, (size_t)NNODES * 64 * 4, stream);
  egnn_edge_kernel<<<NEDGES / 64, 128, 0, stream>>>(
      ei, feats1, cB, ew1p + 64 * 128, eb1 + 114, ew2p + 128 * 64, eb2 + 64,
      cw1p + 64 * 256, cb1 + 256, cw2 + 256, cb2 + 1, m_i, cB, 0);
  node_kernel<<<(NNODES + 255) / 256, 256, 0, stream>>>(
      feats1, m_i, nw1 + 88 * 48, nb1 + 48, nw2 + 48 * 24, nb2 + 24, feats2);

  // ---- FFNN + pooling + head ----
  count_kernel<<<(NNODES + 255) / 256, 256, 0, stream>>>(bidx, cnt);
  const int ntiles = (NNODES + 15) / 16;  // 3125
  ffnn_kernel<<<(ntiles + 1) / 2, 64, 0, stream>>>(feats0, feats1, feats2, bidx,
                                                   f0wp, f0b, f1wp, f1b, f2wp, f2b,
                                                   pooled, ntiles);
  final_kernel<<<1, 256, 0, stream>>>(pooled, cnt, fow, fob, (float*)d_out);

  (void)in_sizes; (void)n_in; (void)out_size; (void)ws_size;
}
